// PatchcoreModel_27608049778785
// MI455X (gfx1250) — compile-verified, matrix operand order checked by reading, not run on hardware
//
#include <hip/hip_runtime.h>

// ---------------------------------------------------------------------------
// PatchCore on MI455X (gfx1250): fused distance-GEMM + top-9.
// bf16 hi/lo split GEMM (3x v_wmma_f32_16x16x32_bf16), TDM-staged B tiles,
// software-pipelined A fragments, register-resident top-9, fused blur.
// ---------------------------------------------------------------------------

#define N_ROWS 25088      // B*FH*FW = 32*28*28
#define M_COLS 16384      // memory bank entries
#define K_DIM  1536
#define MT     128        // rows per workgroup
#define NT     128        // cols per tile
#define KT     32         // bf16 WMMA K
#define NTILES (M_COLS / NT)          // 128
#define KSTEPS (K_DIM / KT)           // 48
#define TILE_BYTES (NT * KT * 2 * 2)  // 16384: 8KB hi + 8KB lo
#define N_TILE_TOTAL (NTILES * KSTEPS)

typedef __attribute__((ext_vector_type(16))) __bf16 bf16x16;
typedef __attribute__((ext_vector_type(8)))  __bf16 bf16x8;
typedef __attribute__((ext_vector_type(8)))  float  f32x8;
typedef __attribute__((ext_vector_type(4)))  unsigned u32x4;
typedef __attribute__((ext_vector_type(4)))  int      i32x4;
typedef __attribute__((ext_vector_type(8)))  int      i32x8;

#if defined(__has_builtin)
#if __has_builtin(__builtin_amdgcn_tensor_load_to_lds)
#define HAVE_TDM 1
#endif
#endif
#ifndef HAVE_TDM
#define HAVE_TDM 0
#endif

__device__ __forceinline__ __bf16 cvt_bf16(float f) {
  unsigned u = __builtin_bit_cast(unsigned, f);
  u += 0x7FFFu + ((u >> 16) & 1u);          // round-to-nearest-even
  unsigned short h = (unsigned short)(u >> 16);
  return __builtin_bit_cast(__bf16, h);
}
__device__ __forceinline__ float bf16_to_f32(__bf16 b) {
  unsigned short h = __builtin_bit_cast(unsigned short, b);
  unsigned u = ((unsigned)h) << 16;
  return __builtin_bit_cast(float, u);
}
__device__ __forceinline__ f32x8 wmma_bf16(bf16x16 a, bf16x16 b, f32x8 c) {
  return __builtin_amdgcn_wmma_f32_16x16x32_bf16(false, a, false, b,
                                                 (short)0, c, false, false);
}

// ---------------------------------------------------------------------------
// Tensor Data Mover: contiguous 16KB tile global -> LDS (1-row 2D descriptor)
// ---------------------------------------------------------------------------
__device__ __forceinline__ void tdm_load_tile(const void* gsrc, void* ldst) {
#if HAVE_TDM
  unsigned long long ga = (unsigned long long)gsrc;
  unsigned la = (unsigned)(unsigned long long)ldst;   // LDS byte address
  u32x4 g0;
  g0[0] = 1u;                                         // count=1, user mode
  g0[1] = la;                                         // lds_addr
  g0[2] = (unsigned)(ga & 0xFFFFFFFFu);               // global_addr[31:0]
  g0[3] = (unsigned)((ga >> 32) & 0x01FFFFFFu) | 0x80000000u;  // addr hi | type=2
  i32x8 g1;
  g1[0] = 0x00030000;     // wg_mask=0, data_size=3 (8B units)
  g1[1] = 0x08000000;     // tensor_dim0 = 2048 (in [63:48])
  g1[2] = 0x00010000;     // tensor_dim1 = 1    (in [111:80] low part)
  g1[3] = 0x08000000;     // tile_dim0 = 2048   (in [127:112])
  g1[4] = 0x00000001;     // tile_dim1 = 1
  g1[5] = 2048;           // tensor_dim0_stride lo
  g1[6] = 0;
  g1[7] = 0;
  i32x4 z4;
  z4[0] = 0; z4[1] = 0; z4[2] = 0; z4[3] = 0;
#if __clang_major__ >= 23
  i32x8 z8;
  z8[0] = 0; z8[1] = 0; z8[2] = 0; z8[3] = 0;
  z8[4] = 0; z8[5] = 0; z8[6] = 0; z8[7] = 0;
  __builtin_amdgcn_tensor_load_to_lds(g0, g1, z4, z4, z8, 0);
#else
  __builtin_amdgcn_tensor_load_to_lds(g0, g1, z4, z4, 0);
#endif
#else
  (void)gsrc; (void)ldst;
#endif
}

__device__ __forceinline__ void stage_tile(void* ldsDst, const void* gSrc,
                                           int tid, int wave) {
#if HAVE_TDM
  if (wave == 0) tdm_load_tile(gSrc, ldsDst);
  (void)tid;
#else
  const uint4* s = (const uint4*)gSrc + tid;          // 1024 x 16B chunks
  uint4* d = (uint4*)ldsDst + tid;
#pragma unroll
  for (int q = 0; q < 4; ++q) d[q * 256] = s[q * 256];
  (void)wave;
#endif
}

__device__ __forceinline__ void stage_wait(int wave) {
#if HAVE_TDM
  if (wave == 0) {
#if __has_builtin(__builtin_amdgcn_s_wait_tensorcnt)
    __builtin_amdgcn_s_wait_tensorcnt((short)0);
#else
    asm volatile("s_wait_tensorcnt 0x0" ::: "memory");
#endif
  }
#else
  (void)wave;
#endif
}

// top-9 insert: static-index min/max bubble, stays in VGPRs
__device__ __forceinline__ void top9_update(float (&t9)[9], float v) {
  if (v < t9[8]) {
    t9[8] = v;
#pragma unroll
    for (int j = 8; j > 0; --j) {
      float lo2 = fminf(t9[j - 1], t9[j]);
      float hi2 = fmaxf(t9[j - 1], t9[j]);
      t9[j - 1] = lo2;
      t9[j]     = hi2;
    }
  }
}

// ---------------------------------------------------------------------------
// row squared-norms
// ---------------------------------------------------------------------------
__global__ void row_norm_kernel(const float* __restrict__ x, int K,
                                float* __restrict__ out) {
  __shared__ float red[256];
  const int row = blockIdx.x;
  const float* p = x + (size_t)row * K;
  float s = 0.f;
  for (int i = threadIdx.x; i < K; i += 256) { float v = p[i]; s += v * v; }
  red[threadIdx.x] = s;
  __syncthreads();
  for (int off = 128; off > 0; off >>= 1) {
    if (threadIdx.x < off) red[threadIdx.x] += red[threadIdx.x + off];
    __syncthreads();
  }
  if (threadIdx.x == 0) out[row] = red[0];
}

// ---------------------------------------------------------------------------
// prep: split memory bank into bf16 hi/lo, tile-contiguous fragment layout
// ---------------------------------------------------------------------------
__global__ void split_bank_kernel(const float* __restrict__ Bm,
                                  char* __restrict__ tiles) {
  const int tid = threadIdx.x;
  const int nt = blockIdx.x / KSTEPS;
  const int ks = blockIdx.x % KSTEPS;
  const int col = tid >> 1;
  const int kh  = tid & 1;
  const float* src = Bm + (size_t)(nt * NT + col) * K_DIM + ks * KT + kh * 16;
  float4 f0 = *(const float4*)(src);
  float4 f1 = *(const float4*)(src + 4);
  float4 f2 = *(const float4*)(src + 8);
  float4 f3 = *(const float4*)(src + 12);
  float fv[16] = {f0.x, f0.y, f0.z, f0.w, f1.x, f1.y, f1.z, f1.w,
                  f2.x, f2.y, f2.z, f2.w, f3.x, f3.y, f3.z, f3.w};
  bf16x8 h0, h1, l0, l1;
#pragma unroll
  for (int e = 0; e < 8; ++e) {
    __bf16 hb = cvt_bf16(fv[e]);
    h0[e] = hb;
    l0[e] = cvt_bf16(fv[e] - bf16_to_f32(hb));
    __bf16 hb2 = cvt_bf16(fv[8 + e]);
    h1[e] = hb2;
    l1[e] = cvt_bf16(fv[8 + e] - bf16_to_f32(hb2));
  }
  char* tb = tiles + (size_t)blockIdx.x * TILE_BYTES;
  __bf16* dh = (__bf16*)(tb) + col * KT + kh * 16;
  __bf16* dl = (__bf16*)(tb + TILE_BYTES / 2) + col * KT + kh * 16;
  *(bf16x8*)(dh)     = h0;
  *(bf16x8*)(dh + 8) = h1;
  *(bf16x8*)(dl)     = l0;
  *(bf16x8*)(dl + 8) = l1;
}

// ---------------------------------------------------------------------------
// prep: split embedding into row-major bf16 hi/lo
// ---------------------------------------------------------------------------
__global__ void split_a_kernel(const float* __restrict__ A,
                               __bf16* __restrict__ Ahi,
                               __bf16* __restrict__ Alo) {
  const size_t idx = ((size_t)blockIdx.x * 256 + threadIdx.x) * 8;
  float4 f0 = *(const float4*)(A + idx);
  float4 f1 = *(const float4*)(A + idx + 4);
  float fv[8] = {f0.x, f0.y, f0.z, f0.w, f1.x, f1.y, f1.z, f1.w};
  bf16x8 h, l;
#pragma unroll
  for (int e = 0; e < 8; ++e) {
    __bf16 hb = cvt_bf16(fv[e]);
    h[e] = hb;
    l[e] = cvt_bf16(fv[e] - bf16_to_f32(hb));
  }
  *(bf16x8*)(Ahi + idx) = h;
  *(bf16x8*)(Alo + idx) = l;
}

// ---------------------------------------------------------------------------
// FAST PATH: pre-split bf16 operands, TDM double-buffered B tiles,
// A fragments software-pipelined one k-step ahead
// ---------------------------------------------------------------------------
__global__ __launch_bounds__(256)
void patch_dist_topk_fast(const __bf16* __restrict__ Ahi,
                          const __bf16* __restrict__ Alo,
                          const char* __restrict__ tiles,
                          const float* __restrict__ x2,
                          const float* __restrict__ y2,
                          float* __restrict__ top9,
                          float* __restrict__ mins) {
  extern __shared__ char smem[];
  float* distS = (float*)smem;                     // 128x128 f32 (64KB)
  char*  bBuf0 = smem + MT * NT * 4;               // 2 x 16KB B tile buffers
  char*  bBuf1 = bBuf0 + TILE_BYTES;

  const int tid  = threadIdx.x;
  const int lane = tid & 31;
  const int wave = tid >> 5;
  const int hh   = lane >> 4;
  const int ln   = lane & 15;
  const int rowBase = blockIdx.x * MT;
  const int wRow    = rowBase + wave * 16;

  float x2v[8];
#pragma unroll
  for (int r = 0; r < 8; ++r) x2v[r] = x2[wRow + hh * 8 + r];

  float t9[9];
#pragma unroll
  for (int j = 0; j < 9; ++j) t9[j] = 3.0e38f;

  const int srow  = tid >> 1;
  const int shalf = tid & 1;

  const __bf16* ahRow = Ahi + (size_t)(wRow + ln) * K_DIM + hh * 8;
  const __bf16* alRow = Alo + (size_t)(wRow + ln) * K_DIM + hh * 8;

  // prologue: DMA first tile into buffer 0
  stage_tile(bBuf0, tiles, tid, wave);

  for (int nt = 0; nt < NTILES; ++nt) {
    const int n0 = nt * NT;
    f32x8 acc[8];
#pragma unroll
    for (int t = 0; t < 8; ++t)
#pragma unroll
      for (int e = 0; e < 8; ++e) acc[t][e] = 0.0f;

    // prime the A-fragment pipeline for ks = 0
    bf16x8 a0c = *(const bf16x8*)(ahRow);
    bf16x8 a1c = *(const bf16x8*)(ahRow + 16);
    bf16x8 a2c = *(const bf16x8*)(alRow);
    bf16x8 a3c = *(const bf16x8*)(alRow + 16);

    for (int ks = 0; ks < KSTEPS; ++ks) {
      const int cur = ks & 1;
      const int ti  = nt * KSTEPS + ks;

      stage_wait(wave);          // current B tile landed (TENSORcnt)
      __syncthreads();           // visible to all waves; prev buffer free

      if (ti + 1 < N_TILE_TOTAL) // DMA next B tile into the other buffer
        stage_tile(cur ? bBuf0 : bBuf1,
                   tiles + (size_t)(ti + 1) * TILE_BYTES, tid, wave);

      // issue A loads for ks+1 now; consumed next iteration (hides L2 latency)
      const int kn = (ks + 1 < KSTEPS) ? (ks + 1) * KT : ks * KT;
      bf16x8 a0n = *(const bf16x8*)(ahRow + kn);
      bf16x8 a1n = *(const bf16x8*)(ahRow + kn + 16);
      bf16x8 a2n = *(const bf16x8*)(alRow + kn);
      bf16x8 a3n = *(const bf16x8*)(alRow + kn + 16);

      bf16x16 aHi = __builtin_shufflevector(a0c, a1c, 0, 1, 2, 3, 4, 5, 6, 7,
                                            8, 9, 10, 11, 12, 13, 14, 15);
      bf16x16 aLo = __builtin_shufflevector(a2c, a3c, 0, 1, 2, 3, 4, 5, 6, 7,
                                            8, 9, 10, 11, 12, 13, 14, 15);

      const __bf16* bufHi = (const __bf16*)(cur ? bBuf1 : bBuf0);
#pragma unroll
      for (int t = 0; t < 8; ++t) {
        const __bf16* pb = bufHi + (t * 16 + ln) * KT + hh * 8;
        bf16x8 c0 = *(const bf16x8*)(pb);
        bf16x8 c1 = *(const bf16x8*)(pb + 16);
        bf16x16 bh = __builtin_shufflevector(c0, c1, 0, 1, 2, 3, 4, 5, 6, 7,
                                             8, 9, 10, 11, 12, 13, 14, 15);
        bf16x8 d0 = *(const bf16x8*)(pb + 4096);        // lo half of tile
        bf16x8 d1 = *(const bf16x8*)(pb + 4096 + 16);
        bf16x16 bl = __builtin_shufflevector(d0, d1, 0, 1, 2, 3, 4, 5, 6, 7,
                                             8, 9, 10, 11, 12, 13, 14, 15);
        acc[t] = wmma_bf16(aHi, bh, acc[t]);
        acc[t] = wmma_bf16(aHi, bl, acc[t]);
        acc[t] = wmma_bf16(aLo, bh, acc[t]);
      }

      a0c = a0n; a1c = a1n; a2c = a2n; a3c = a3n;   // advance pipeline
    }

    // epilogue: d = sqrt(x2 + y2 - 2*s) into LDS tile
    __syncthreads();
#pragma unroll
    for (int t = 0; t < 8; ++t) {
      const int col = n0 + t * 16 + ln;
      const float y2v = y2[col];
#pragma unroll
      for (int r = 0; r < 8; ++r) {
        float d2 = x2v[r] + y2v - 2.0f * acc[t][r];
        float d  = sqrtf(fmaxf(d2, 1e-12f));
        distS[(wave * 16 + hh * 8 + r) * NT + (t * 16 + ln)] = d;
      }
    }
    __syncthreads();

    // vectorized top-9 scan: 16 x ds_load_b128 per tile per thread
    const float4* rp = (const float4*)(distS + srow * NT + shalf * 64);
#pragma unroll 4
    for (int c = 0; c < 16; ++c) {
      float4 v = rp[c];
      top9_update(t9, v.x);
      top9_update(t9, v.y);
      top9_update(t9, v.z);
      top9_update(t9, v.w);
    }
    __syncthreads();
  }

  // merge the two half-row sorted lists
  float* msh = distS;
#pragma unroll
  for (int j = 0; j < 9; ++j) msh[tid * 9 + j] = t9[j];
  __syncthreads();
  if (shalf == 0) {
    const float* a = msh + tid * 9;
    const float* b = msh + (tid + 1) * 9;
    int i = 0, j = 0;
    const int grow = rowBase + srow;
    float first = 0.f;
    for (int k = 0; k < 9; ++k) {
      float avv = (i < 9) ? a[i] : 3.0e38f;
      float bvv = (j < 9) ? b[j] : 3.0e38f;
      float pick;
      if (avv <= bvv) { pick = avv; ++i; } else { pick = bvv; ++j; }
      if (k == 0) first = pick;
      top9[(size_t)grow * 9 + k] = pick;
    }
    mins[grow] = first;
  }
}

// ---------------------------------------------------------------------------
// FALLBACK PATH: convert-on-the-fly (small workspace)
// ---------------------------------------------------------------------------
__global__ __launch_bounds__(256)
void patch_dist_topk_fb(const float* __restrict__ A,
                        const float* __restrict__ Bm,
                        const float* __restrict__ x2,
                        const float* __restrict__ y2,
                        float* __restrict__ top9,
                        float* __restrict__ mins) {
  extern __shared__ char smem[];
  float*  distS = (float*)smem;
  __bf16* bHiS  = (__bf16*)(smem + MT * NT * 4);
  __bf16* bLoS  = (__bf16*)(smem + MT * NT * 4 + NT * KT * 2);

  const int tid  = threadIdx.x;
  const int lane = tid & 31;
  const int wave = tid >> 5;
  const int hh   = lane >> 4;
  const int ln   = lane & 15;
  const int rowBase = blockIdx.x * MT;
  const int wRow    = rowBase + wave * 16;

  float x2v[8];
#pragma unroll
  for (int r = 0; r < 8; ++r) x2v[r] = x2[wRow + hh * 8 + r];

  float t9[9];
#pragma unroll
  for (int j = 0; j < 9; ++j) t9[j] = 3.0e38f;

  const int srow  = tid >> 1;
  const int shalf = tid & 1;
  const float* arowBase = A + (size_t)(wRow + ln) * K_DIM;

  for (int n0 = 0; n0 < M_COLS; n0 += NT) {
    f32x8 acc[8];
#pragma unroll
    for (int t = 0; t < 8; ++t)
#pragma unroll
      for (int e = 0; e < 8; ++e) acc[t][e] = 0.0f;

    for (int k0 = 0; k0 < K_DIM; k0 += KT) {
      {
        const int bcol = tid >> 1;
        const int kh   = tid & 1;
        const float* src = Bm + (size_t)(n0 + bcol) * K_DIM + k0 + kh * 16;
        float4 f0 = *(const float4*)(src);
        float4 f1 = *(const float4*)(src + 4);
        float4 f2 = *(const float4*)(src + 8);
        float4 f3 = *(const float4*)(src + 12);
        float fv[16] = {f0.x, f0.y, f0.z, f0.w, f1.x, f1.y, f1.z, f1.w,
                        f2.x, f2.y, f2.z, f2.w, f3.x, f3.y, f3.z, f3.w};
        bf16x8 h0, h1, l0, l1;
#pragma unroll
        for (int e = 0; e < 8; ++e) {
          __bf16 hb = cvt_bf16(fv[e]);
          h0[e] = hb;
          l0[e] = cvt_bf16(fv[e] - bf16_to_f32(hb));
          __bf16 hb2 = cvt_bf16(fv[8 + e]);
          h1[e] = hb2;
          l1[e] = cvt_bf16(fv[8 + e] - bf16_to_f32(hb2));
        }
        __bf16* dh = bHiS + bcol * KT + kh * 16;
        __bf16* dl = bLoS + bcol * KT + kh * 16;
        *(bf16x8*)(dh)     = h0;
        *(bf16x8*)(dh + 8) = h1;
        *(bf16x8*)(dl)     = l0;
        *(bf16x8*)(dl + 8) = l1;
      }
      __syncthreads();

      const float* ar = arowBase + k0 + hh * 8;
      float4 a0 = *(const float4*)(ar);
      float4 a1 = *(const float4*)(ar + 4);
      float4 a2 = *(const float4*)(ar + 16);
      float4 a3 = *(const float4*)(ar + 20);
      float av[16] = {a0.x, a0.y, a0.z, a0.w, a1.x, a1.y, a1.z, a1.w,
                      a2.x, a2.y, a2.z, a2.w, a3.x, a3.y, a3.z, a3.w};
      bf16x16 aHi, aLo;
#pragma unroll
      for (int e = 0; e < 16; ++e) {
        __bf16 hb = cvt_bf16(av[e]);
        aHi[e] = hb;
        aLo[e] = cvt_bf16(av[e] - bf16_to_f32(hb));
      }

#pragma unroll
      for (int t = 0; t < 8; ++t) {
        const __bf16* pb = bHiS + (t * 16 + ln) * KT + hh * 8;
        bf16x8 c0 = *(const bf16x8*)(pb);
        bf16x8 c1 = *(const bf16x8*)(pb + 16);
        bf16x16 bh = __builtin_shufflevector(c0, c1, 0, 1, 2, 3, 4, 5, 6, 7,
                                             8, 9, 10, 11, 12, 13, 14, 15);
        const __bf16* pl = bLoS + (t * 16 + ln) * KT + hh * 8;
        bf16x8 d0 = *(const bf16x8*)(pl);
        bf16x8 d1 = *(const bf16x8*)(pl + 16);
        bf16x16 bl = __builtin_shufflevector(d0, d1, 0, 1, 2, 3, 4, 5, 6, 7,
                                             8, 9, 10, 11, 12, 13, 14, 15);
        acc[t] = wmma_bf16(aHi, bh, acc[t]);
        acc[t] = wmma_bf16(aHi, bl, acc[t]);
        acc[t] = wmma_bf16(aLo, bh, acc[t]);
      }
      __syncthreads();
    }

#pragma unroll
    for (int t = 0; t < 8; ++t) {
      const int col = n0 + t * 16 + ln;
      const float y2v = y2[col];
#pragma unroll
      for (int r = 0; r < 8; ++r) {
        float d2 = x2v[r] + y2v - 2.0f * acc[t][r];
        float d  = sqrtf(fmaxf(d2, 1e-12f));
        distS[(wave * 16 + hh * 8 + r) * NT + (t * 16 + ln)] = d;
      }
    }
    __syncthreads();

    const float4* rp = (const float4*)(distS + srow * NT + shalf * 64);
#pragma unroll 4
    for (int c = 0; c < 16; ++c) {
      float4 v = rp[c];
      top9_update(t9, v.x);
      top9_update(t9, v.y);
      top9_update(t9, v.z);
      top9_update(t9, v.w);
    }
    __syncthreads();
  }

  float* msh = distS;
#pragma unroll
  for (int j = 0; j < 9; ++j) msh[tid * 9 + j] = t9[j];
  __syncthreads();
  if (shalf == 0) {
    const float* a = msh + tid * 9;
    const float* b = msh + (tid + 1) * 9;
    int i = 0, j = 0;
    const int grow = rowBase + srow;
    float first = 0.f;
    for (int k = 0; k < 9; ++k) {
      float avv = (i < 9) ? a[i] : 3.0e38f;
      float bvv = (j < 9) ? b[j] : 3.0e38f;
      float pick;
      if (avv <= bvv) { pick = avv; ++i; } else { pick = bvv; ++j; }
      if (k == 0) first = pick;
      top9[(size_t)grow * 9 + k] = pick;
    }
    mins[grow] = first;
  }
}

// ---------------------------------------------------------------------------
// anomaly map: nearest 8x upsample fused into separable 33-tap gaussian blur
// ---------------------------------------------------------------------------
__device__ __forceinline__ int reflect224(int p) {
  p = (p < 0) ? -p : p;
  return (p > 223) ? (446 - p) : p;
}

__global__ void blur_h_kernel(const float* __restrict__ mins,
                              float* __restrict__ tmp) {
  __shared__ float g[33];
  __shared__ float norm;
  const int tid = threadIdx.x;
  if (tid < 33) {
    float a = (float)tid - 16.0f;
    g[tid] = __expf(-0.03125f * a * a);    // sigma = 4
  }
  __syncthreads();
  if (tid == 0) {
    float s = 0.f;
    for (int i = 0; i < 33; ++i) s += g[i];
    norm = 1.0f / s;
  }
  __syncthreads();
  const int b = blockIdx.x / 224;
  const int y = blockIdx.x % 224;
  if (tid < 224) {
    const int x = tid;
    const float* mrow = mins + b * 784 + (y >> 3) * 28;
    float acc = 0.f;
#pragma unroll
    for (int i = 0; i < 33; ++i) {
      int xx = reflect224(x - 16 + i);
      acc += g[i] * mrow[xx >> 3];
    }
    tmp[(size_t)b * 50176 + y * 224 + x] = acc * norm;
  }
}

__global__ void blur_v_kernel(const float* __restrict__ tmp,
                              float* __restrict__ out) {
  __shared__ float g[33];
  __shared__ float norm;
  const int tid = threadIdx.x;
  if (tid < 33) {
    float a = (float)tid - 16.0f;
    g[tid] = __expf(-0.03125f * a * a);
  }
  __syncthreads();
  if (tid == 0) {
    float s = 0.f;
    for (int i = 0; i < 33; ++i) s += g[i];
    norm = 1.0f / s;
  }
  __syncthreads();
  const int b = blockIdx.x / 224;
  const int x = blockIdx.x % 224;
  if (tid < 224) {
    const int y = tid;
    float acc = 0.f;
#pragma unroll
    for (int i = 0; i < 33; ++i) {
      int yy = reflect224(y - 16 + i);
      acc += g[i] * tmp[(size_t)b * 50176 + yy * 224 + x];
    }
    out[(size_t)b * 50176 + y * 224 + x] = acc * norm;
  }
}

// ---------------------------------------------------------------------------
// image-level score
// ---------------------------------------------------------------------------
__global__ void score_kernel(const float* __restrict__ mins,
                             const float* __restrict__ top9,
                             float* __restrict__ outScore) {
  __shared__ float sv[256];
  __shared__ int   si[256];
  const int tid = threadIdx.x;
  float best = -1e30f;
  int   bidx = 0;
  for (int i = tid; i < N_ROWS; i += 256) {
    float v = mins[i];
    if (v > best) { best = v; bidx = i; }
  }
  sv[tid] = best;
  si[tid] = bidx;
  __syncthreads();
  for (int o = 128; o > 0; o >>= 1) {
    if (tid < o && sv[tid + o] > sv[tid]) { sv[tid] = sv[tid + o]; si[tid] = si[tid + o]; }
    __syncthreads();
  }
  if (tid == 0) {
    const float* t = top9 + (size_t)si[0] * 9;
    const float mx = t[8];
    float se = 0.f;
    for (int j = 0; j < 9; ++j) se += __expf(t[j] - mx);
    float w = 1.0f - 1.0f / se;
    outScore[0] = w * t[0];
  }
}

// ---------------------------------------------------------------------------
extern "C" void kernel_launch(void* const* d_in, const int* in_sizes, int n_in,
                              void* d_out, int out_size, void* d_ws, size_t ws_size,
                              hipStream_t stream) {
  (void)in_sizes; (void)n_in; (void)out_size;
  const float* emb = (const float*)d_in[0];   // [25088, 1536]
  const float* mem = (const float*)d_in[1];   // [16384, 1536]
  float* out = (float*)d_out;

  char* ws = (char*)d_ws;
  size_t off = 0;
  auto carve = [&](size_t bytes) -> void* {
    void* p = ws + off;
    off = (off + bytes + 255) & ~(size_t)255;
    return p;
  };
  float* x2   = (float*)carve((size_t)N_ROWS * 4);
  float* y2   = (float*)carve((size_t)M_COLS * 4);
  float* top9 = (float*)carve((size_t)N_ROWS * 9 * 4);
  float* mins = (float*)carve((size_t)N_ROWS * 4);
  float* btmp = (float*)carve((size_t)32 * 224 * 224 * 4);

  char*   bankTiles = (char*)carve((size_t)N_TILE_TOTAL * TILE_BYTES);  // ~100.7MB
  __bf16* Ahi = (__bf16*)carve((size_t)N_ROWS * K_DIM * 2);             // ~77MB
  __bf16* Alo = (__bf16*)carve((size_t)N_ROWS * K_DIM * 2);             // ~77MB
  const bool fastOK = (ws_size >= off);

  row_norm_kernel<<<N_ROWS, 256, 0, stream>>>(emb, K_DIM, x2);
  row_norm_kernel<<<M_COLS, 256, 0, stream>>>(mem, K_DIM, y2);

  if (fastOK) {
    split_bank_kernel<<<N_TILE_TOTAL, 256, 0, stream>>>(mem, bankTiles);
    split_a_kernel<<<(N_ROWS * K_DIM) / (256 * 8), 256, 0, stream>>>(emb, Ahi, Alo);
    const size_t smemFast = MT * NT * 4 + 2 * TILE_BYTES;   // 64KB + 32KB
    patch_dist_topk_fast<<<N_ROWS / MT, 256, smemFast, stream>>>(
        Ahi, Alo, bankTiles, x2, y2, top9, mins);
  } else {
    const size_t smemFb = MT * NT * 4 + 2 * (NT * KT * 2);  // 64KB + 16KB
    patch_dist_topk_fb<<<N_ROWS / MT, 256, smemFb, stream>>>(
        emb, mem, x2, y2, top9, mins);
  }

  blur_h_kernel<<<32 * 224, 256, 0, stream>>>(mins, btmp);
  blur_v_kernel<<<32 * 224, 256, 0, stream>>>(btmp, out);
  score_kernel<<<1, 256, 0, stream>>>(mins, top9, out + 32 * 224 * 224);
}
